// GATImageClassifier_24446953848871
// MI455X (gfx1250) — matrix operand order repaired and checked by reading, not compiled, not run on hardware
//
#include <hip/hip_runtime.h>
#include <math.h>

// ---------------- problem constants (match reference) ----------------
#define NN      20000
#define EE      320000
#define GG      64
#define EDGES   (EE + NN)        // 340000 (self-loops appended)
#define IN_DIM  128
#define HID     256
#define HEADS   4
#define CPH     64               // channels per head
#define NCLS    100

typedef __attribute__((ext_vector_type(16))) _Float16 v16h;
typedef __attribute__((ext_vector_type(8)))  float    v8f;

// ---------------- helpers ----------------
__device__ __forceinline__ unsigned f2mono(float x) {
  unsigned u = __float_as_uint(x);
  return (u & 0x80000000u) ? ~u : (u | 0x80000000u);
}
__device__ __forceinline__ float mono2f(unsigned u) {
  return (u & 0x80000000u) ? __uint_as_float(u ^ 0x80000000u) : __uint_as_float(~u);
}
__device__ __forceinline__ void edge_nodes(const long long* __restrict__ ei, int e,
                                           int& s, int& d) {
  if (e < EE) { s = (int)ei[e]; d = (int)ei[EE + e]; }
  else        { s = d = e - EE; }          // appended self-loops
}

// ---------------- fills ----------------
__global__ void fill_f32(float* p, float v, int n) {
  int i = blockIdx.x * blockDim.x + threadIdx.x;
  if (i < n) p[i] = v;
}
__global__ void fill_u32(unsigned* p, unsigned v, int n) {
  int i = blockIdx.x * blockDim.x + threadIdx.x;
  if (i < n) p[i] = v;
}

// ------------- weight pre-pack: f32 row-major [K,Ncol] -> f16 B-fragments ----
// Fragment f = kt*(Ncol/16)+nt holds 32 lanes x 16 halves, contiguous per lane.
// Lane layout (contiguous k order, not the ISA table order): lanes 0-15 K=kt*32..+15,
// lanes 16-31 K=kt*32+16..+31; element i = row (kt*32 + (lane/16)*16 + i), col n.
__global__ void pack_w_kernel(const float* __restrict__ W,
                              _Float16* __restrict__ out, int K, int Ncol) {
  int idx = blockIdx.x * blockDim.x + threadIdx.x;
  int ntiles = Ncol >> 4;
  int total = (K >> 5) * ntiles * 32;
  if (idx >= total) return;
  int lane = idx & 31;
  int f = idx >> 5;
  int kt = f / ntiles, nt = f % ntiles;
  int n = (nt << 4) + (lane & 15);
  _Float16* o = out + (size_t)f * 512 + lane * 16;
  const float* src = W + (size_t)((kt << 5) + ((lane >> 4) << 4)) * Ncol + n;
#pragma unroll
  for (int i = 0; i < 16; ++i) o[i] = (_Float16)src[(size_t)i * Ncol];
}

// ------------- WMMA GEMM: D[M,Ncol] = act(A[M,K] @ B + bias) -----------------
// block = 256 threads (8 waves). Block covers 16 rows x 128 cols
// (grid.y = Ncol/128). A strip staged in LDS as f16 in A-fragment layout;
// B comes pre-packed. Hot loop: 2x ds_load_b128 + 2x global_load_b128 + 1 wmma.
__global__ void gemm_wmma_kernel(const float* __restrict__ A,
                                 const _Float16* __restrict__ Bp,
                                 const float* __restrict__ bias,
                                 float* __restrict__ D,
                                 int M, int K, int Ncol, int act) {
  __shared__ alignas(32) _Float16 sA[16 * 256];   // up to K=256: 8KB
  const int tid    = threadIdx.x;
  const int lane   = tid & 31;
  const int wave   = tid >> 5;
  const int mBase  = blockIdx.x * 16;
  const int ntiles = Ncol >> 4;
  const int ntile  = blockIdx.y * 8 + wave;

  // cooperative stage: A[mBase..mBase+15, 0..K) -> f16 fragment layout in LDS
  // idx = kt*512 + l*16 + i ; lane l holds row m=l%16,
  // K-local = (l/16)*16 + i   (contiguous k order, matches packed B)
  const int total = (K >> 5) * 512;
  for (int idx = tid; idx < total; idx += 256) {
    int kt = idx >> 9;
    int r  = idx & 511;
    int l  = r >> 4;
    int i  = r & 15;
    int m  = l & 15;
    int kl = ((l >> 4) << 4) + i;
    sA[idx] = (_Float16)A[(size_t)(mBase + m) * K + (kt << 5) + kl];
  }
  __syncthreads();

  v8f acc = {};
  for (int kt = 0; kt < (K >> 5); ++kt) {
    v16h a = *(const v16h*)(sA + (kt << 9) + (lane << 4));
    v16h b = *(const v16h*)(Bp + (((size_t)kt * ntiles + ntile) * 32 + lane) * 16);
    acc = __builtin_amdgcn_wmma_f32_16x16x32_f16(false, a, false, b,
                                                 (short)0, acc, false, false);
  }

  // D layout: vgpr j -> row j + 8*(lane>=16), col = lane%16
  const int n    = (ntile << 4) + (lane & 15);
  const int mOff = (lane >> 4) * 8;
  const float bv = bias ? bias[n] : 0.0f;
#pragma unroll
  for (int j = 0; j < 8; ++j) {
    float v = acc[j] + bv;
    if (act) v = fmaxf(v, 0.0f);
    D[(size_t)(mBase + mOff + j) * Ncol + n] = v;
  }
}

// ---------------- attention logits: al_s/al_d [N, HEADS] ----------------
__global__ void attn_logits_kernel(const float* __restrict__ H,
                                   const float* __restrict__ a_s,
                                   const float* __restrict__ a_d,
                                   float* __restrict__ ALS,
                                   float* __restrict__ ALD) {
  int idx = blockIdx.x * blockDim.x + threadIdx.x;
  if (idx >= NN * HEADS) return;
  const int n = idx >> 2, h = idx & 3;
  const float* row = H + (size_t)n * HID + h * CPH;
  const float* as  = a_s + h * CPH;
  const float* ad  = a_d + h * CPH;
  float ss = 0.f, sd = 0.f;
#pragma unroll 8
  for (int c = 0; c < CPH; ++c) { float v = row[c]; ss += v * as[c]; sd += v * ad[c]; }
  ALS[idx] = ss; ALD[idx] = sd;
}

// ---------------- edge pipeline ----------------
__global__ void edge_max_kernel(const long long* __restrict__ ei,
                                const float* __restrict__ ALS,
                                const float* __restrict__ ALD,
                                unsigned* __restrict__ SMAX) {
  int e = blockIdx.x * blockDim.x + threadIdx.x;
  if (e >= EDGES) return;
  int s, d; edge_nodes(ei, e, s, d);
#pragma unroll
  for (int h = 0; h < HEADS; ++h) {
    float v = ALS[s * HEADS + h] + ALD[d * HEADS + h];
    v = v > 0.f ? v : 0.2f * v;                    // leaky_relu(0.2)
    atomicMax(SMAX + d * HEADS + h, f2mono(v));
  }
}

__global__ void edge_exp_kernel(const long long* __restrict__ ei,
                                const float* __restrict__ ALS,
                                const float* __restrict__ ALD,
                                const unsigned* __restrict__ SMAX,
                                float* __restrict__ SSUM,
                                float* __restrict__ alpha) {
  int e = blockIdx.x * blockDim.x + threadIdx.x;
  if (e >= EDGES) return;
  int s, d; edge_nodes(ei, e, s, d);
#pragma unroll
  for (int h = 0; h < HEADS; ++h) {
    float v = ALS[s * HEADS + h] + ALD[d * HEADS + h];
    v = v > 0.f ? v : 0.2f * v;
    float ex = __expf(v - mono2f(SMAX[d * HEADS + h]));
    alpha[(size_t)e * HEADS + h] = ex;
    atomicAdd(SSUM + d * HEADS + h, ex);
  }
}

__global__ void alpha_norm_kernel(const long long* __restrict__ ei,
                                  const float* __restrict__ SSUM,
                                  float* __restrict__ alpha) {
  int idx = blockIdx.x * blockDim.x + threadIdx.x;
  if (idx >= EDGES * HEADS) return;
  int e = idx >> 2, h = idx & 3;
  int s, d; edge_nodes(ei, e, s, d);
  alpha[idx] = alpha[idx] / SSUM[d * HEADS + h];
}

// one block (256 threads) per edge: thread t handles feature t = h*64+c
__global__ void edge_msg_kernel(const long long* __restrict__ ei,
                                const float* __restrict__ alpha,
                                const float* __restrict__ H,
                                float* __restrict__ xout) {
  int e = blockIdx.x;
  int t = threadIdx.x;
  int s, d; edge_nodes(ei, e, s, d);
  int h = t >> 6;
  float al = alpha[(size_t)e * HEADS + h];
  float v  = H[(size_t)s * HID + t] * al;
  atomicAdd(xout + (size_t)d * HID + t, v);
}

// out = elu(acc + bias) + residual
__global__ void finalize_kernel(float* __restrict__ xout,
                                const float* __restrict__ bias,
                                const float* __restrict__ resid, int n) {
  int i = blockIdx.x * blockDim.x + threadIdx.x;
  if (i >= n) return;
  float v = xout[i] + bias[i & (HID - 1)];
  v = v > 0.f ? v : expm1f(v);
  xout[i] = v + resid[i];
}

// ---------------- global attention pooling ----------------
__global__ void gate2_kernel(const float* __restrict__ gate1,
                             const float* __restrict__ w2,
                             const float* __restrict__ b2,
                             const long long* __restrict__ batch,
                             float* __restrict__ G2,
                             unsigned* __restrict__ GMX) {
  int n = blockIdx.x * blockDim.x + threadIdx.x;
  if (n >= NN) return;
  const float* r = gate1 + (size_t)n * 128;
  float s = 0.f;
#pragma unroll 8
  for (int j = 0; j < 128; ++j) s += r[j] * w2[j];
  s += b2[0];
  G2[n] = s;
  atomicMax(GMX + (int)batch[n], f2mono(s));
}

__global__ void gexp_kernel(float* __restrict__ G2,
                            const long long* __restrict__ batch,
                            const unsigned* __restrict__ GMX,
                            float* __restrict__ GSM) {
  int n = blockIdx.x * blockDim.x + threadIdx.x;
  if (n >= NN) return;
  int b = (int)batch[n];
  float ge = __expf(G2[n] - mono2f(GMX[b]));
  G2[n] = ge;
  atomicAdd(GSM + b, ge);
}

__global__ void gpool_kernel(const float* __restrict__ G2,
                             const float* __restrict__ GSM,
                             const long long* __restrict__ batch,
                             const float* __restrict__ xf,
                             float* __restrict__ GEMB) {
  int i = blockIdx.x * blockDim.x + threadIdx.x;
  if (i >= NN * HID) return;
  int n = i >> 8;
  int c = i & (HID - 1);
  int b = (int)batch[n];
  float w = G2[n] / GSM[b];
  atomicAdd(GEMB + (size_t)b * HID + c, w * xf[i]);
}

// ---------------- classifier head 128 -> 100 ----------------
__global__ void logits_kernel(const float* __restrict__ CLS2,
                              const float* __restrict__ w3,
                              const float* __restrict__ b3,
                              float* __restrict__ out) {
  int i = blockIdx.x * blockDim.x + threadIdx.x;
  if (i >= GG * NCLS) return;
  int g = i / NCLS, c = i % NCLS;
  const float* r = CLS2 + (size_t)g * 128;
  float s = 0.f;
#pragma unroll 8
  for (int j = 0; j < 128; ++j) s += r[j] * w3[j * NCLS + c];
  out[i] = s + b3[c];
}

// ---------------- host orchestration ----------------
static void pack_w(const float* W, _Float16* out, int K, int Ncol, hipStream_t st) {
  int total = (K >> 5) * (Ncol >> 4) * 32;
  pack_w_kernel<<<(total + 255) / 256, 256, 0, st>>>(W, out, K, Ncol);
}

static void gemm(const float* A, const _Float16* Bp, const float* bias, float* D,
                 int M, int K, int Ncol, int act, hipStream_t st) {
  gemm_wmma_kernel<<<dim3(M / 16, Ncol / 128), 256, 0, st>>>(A, Bp, bias, D, M, K, Ncol, act);
}

static void run_gat_layer(const float* xin, int K, const _Float16* Wp,
                          const float* a_s, const float* a_d, const float* bias,
                          const float* resid, float* alpha_out, float* xout,
                          float* H, float* ALS, float* ALD,
                          unsigned* SMAX, float* SSUM,
                          const long long* ei, hipStream_t stream) {
  gemm(xin, Wp, nullptr, H, NN, K, HID, 0, stream);
  attn_logits_kernel<<<(NN * HEADS + 255) / 256, 256, 0, stream>>>(H, a_s, a_d, ALS, ALD);
  fill_u32<<<(NN * HEADS + 255) / 256, 256, 0, stream>>>(SMAX, 0u, NN * HEADS);
  fill_f32<<<(NN * HEADS + 255) / 256, 256, 0, stream>>>(SSUM, 0.f, NN * HEADS);
  fill_f32<<<(NN * HID + 255) / 256, 256, 0, stream>>>(xout, 0.f, NN * HID);
  edge_max_kernel<<<(EDGES + 255) / 256, 256, 0, stream>>>(ei, ALS, ALD, SMAX);
  edge_exp_kernel<<<(EDGES + 255) / 256, 256, 0, stream>>>(ei, ALS, ALD, SMAX, SSUM, alpha_out);
  alpha_norm_kernel<<<(EDGES * HEADS + 255) / 256, 256, 0, stream>>>(ei, SSUM, alpha_out);
  edge_msg_kernel<<<EDGES, HID, 0, stream>>>(ei, alpha_out, H, xout);
  finalize_kernel<<<(NN * HID + 255) / 256, 256, 0, stream>>>(xout, bias, resid, NN * HID);
}

extern "C" void kernel_launch(void* const* d_in, const int* in_sizes, int n_in,
                              void* d_out_v, int out_size, void* d_ws, size_t ws_size,
                              hipStream_t stream) {
  // inputs (setup_inputs dict order; params dict flattened in insertion order)
  const float*     x      = (const float*)d_in[0];
  const long long* ei     = (const long long*)d_in[1];
  const long long* batch  = (const long long*)d_in[2];
  const float* g1_W  = (const float*)d_in[3];
  const float* g1_as = (const float*)d_in[4];
  const float* g1_ad = (const float*)d_in[5];
  const float* g1_b  = (const float*)d_in[6];
  const float* res_w = (const float*)d_in[7];
  const float* res_b = (const float*)d_in[8];
  const float* g2_W  = (const float*)d_in[9];
  const float* g2_as = (const float*)d_in[10];
  const float* g2_ad = (const float*)d_in[11];
  const float* g2_b  = (const float*)d_in[12];
  const float* g3_W  = (const float*)d_in[13];
  const float* g3_as = (const float*)d_in[14];
  const float* g3_ad = (const float*)d_in[15];
  const float* g3_b  = (const float*)d_in[16];
  const float* gw1   = (const float*)d_in[17];
  const float* gb1   = (const float*)d_in[18];
  const float* gw2   = (const float*)d_in[19];
  const float* gb2   = (const float*)d_in[20];
  const float* cw1   = (const float*)d_in[21];
  const float* cb1   = (const float*)d_in[22];
  const float* cw2   = (const float*)d_in[23];
  const float* cb2   = (const float*)d_in[24];
  const float* cw3   = (const float*)d_in[25];
  const float* cb3   = (const float*)d_in[26];

  // output slices: logits | xf | att1 | att2
  float* out    = (float*)d_out_v;
  float* LOGITS = out;                                  // GG*NCLS
  float* XF     = LOGITS + (size_t)GG * NCLS;           // NN*HID
  float* ATT1   = XF + (size_t)NN * HID;                // EDGES*HEADS
  float* ATT2   = ATT1 + (size_t)EDGES * HEADS;         // EDGES*HEADS

  // workspace layout (floats; every region is 32B-aligned by construction)
  float* w    = (float*)d_ws;
  float* H    = w;                         w += (size_t)NN * HID;
  float* XA   = w;                         w += (size_t)NN * HID;
  float* XB   = w;                         w += (size_t)NN * HID;
  float* RES  = w;                         w += (size_t)NN * HID;   // also gate1 [NN,128]
  float* ALS  = w;                         w += (size_t)NN * HEADS;
  float* ALD  = w;                         w += (size_t)NN * HEADS;
  unsigned* SMAX = (unsigned*)w;           w += (size_t)NN * HEADS;
  float* SSUM = w;                         w += (size_t)NN * HEADS;
  float* A3   = w;                         w += (size_t)EDGES * HEADS;
  float* G2   = w;                         w += (size_t)NN;
  unsigned* GMX = (unsigned*)w;            w += (size_t)GG;
  float* GSM  = w;                         w += (size_t)GG;
  float* GEMB = w;                         w += (size_t)GG * HID;
  float* CLS1 = w;                         w += (size_t)GG * HID;
  float* CLS2 = w;                         w += (size_t)GG * 128;
  // packed f16 weight fragments
  _Float16* WP   = (_Float16*)w;
  _Float16* g1p  = WP;                     WP += (size_t)IN_DIM * HID;
  _Float16* resp = WP;                     WP += (size_t)IN_DIM * HID;
  _Float16* g2p  = WP;                     WP += (size_t)HID * HID;
  _Float16* g3p  = WP;                     WP += (size_t)HID * HID;
  _Float16* gw1p = WP;                     WP += (size_t)HID * 128;
  _Float16* cw1p = WP;                     WP += (size_t)HID * HID;
  _Float16* cw2p = WP;                     WP += (size_t)HID * 128;

  // --- pre-pack all GEMM weights to f16 B-fragments ---
  pack_w(g1_W,  g1p,  IN_DIM, HID, stream);
  pack_w(res_w, resp, IN_DIM, HID, stream);
  pack_w(g2_W,  g2p,  HID,    HID, stream);
  pack_w(g3_W,  g3p,  HID,    HID, stream);
  pack_w(gw1,   gw1p, HID,    128, stream);
  pack_w(cw1,   cw1p, HID,    HID, stream);
  pack_w(cw2,   cw2p, HID,    128, stream);

  // --- layer 1 (residual projection) ---
  gemm(x, resp, res_b, RES, NN, IN_DIM, HID, 0, stream);
  run_gat_layer(x, IN_DIM, g1p, g1_as, g1_ad, g1_b, RES, ATT1, XA,
                H, ALS, ALD, SMAX, SSUM, ei, stream);
  // --- layer 2 (identity residual) ---
  run_gat_layer(XA, HID, g2p, g2_as, g2_ad, g2_b, XA, ATT2, XB,
                H, ALS, ALD, SMAX, SSUM, ei, stream);
  // --- layer 3 (writes xf directly into d_out) ---
  run_gat_layer(XB, HID, g3p, g3_as, g3_ad, g3_b, XB, A3, XF,
                H, ALS, ALD, SMAX, SSUM, ei, stream);

  // --- global attention pooling ---
  gemm(XF, gw1p, gb1, RES, NN, HID, 128, 1, stream);
  fill_u32<<<1, 256, 0, stream>>>(GMX, 0u, GG);
  fill_f32<<<1, 256, 0, stream>>>(GSM, 0.f, GG);
  fill_f32<<<(GG * HID + 255) / 256, 256, 0, stream>>>(GEMB, 0.f, GG * HID);
  gate2_kernel<<<(NN + 255) / 256, 256, 0, stream>>>(RES, gw2, gb2, batch, G2, GMX);
  gexp_kernel<<<(NN + 255) / 256, 256, 0, stream>>>(G2, batch, GMX, GSM);
  gpool_kernel<<<(NN * HID + 255) / 256, 256, 0, stream>>>(G2, GSM, batch, XF, GEMB);

  // --- classifier ---
  gemm(GEMB, cw1p, cb1, CLS1, GG, HID, HID, 1, stream);
  gemm(CLS1, cw2p, cb2, CLS2, GG, HID, 128, 1, stream);
  logits_kernel<<<(GG * NCLS + 255) / 256, 256, 0, stream>>>(CLS2, cw3, cb3, LOGITS);
}
